// KnowledgeEnhancedSeq2Seq_55113020342410
// MI455X (gfx1250) — matrix operand order repaired and checked by reading, not compiled, not run on hardware
//
#include <hip/hip_runtime.h>
#include <hip/hip_bf16.h>

typedef __attribute__((ext_vector_type(16))) _Float16 v16h;
typedef __attribute__((ext_vector_type(8)))  float    v8f;

#define BDIM 16      // batch
#define TTs  8       // topic steps
#define Ts   64      // essay steps
#define Ls   100     // memory slots
#define Vs   32000   // vocab
#define Es   512
#define Hs   512
#define EOs  1024
#define As   512
#define INs  2048
#define TEMPs 1024

__device__ __forceinline__ float sigm(float x) { return 1.0f / (1.0f + __expf(-x)); }

// ---- WMMA fragment loaders (layouts per CDNA5 ISA 7.12.2, 16-bit A 16x32 / B 32x16) ----
// A: lanes 0-15 -> rows m0..m0+15, K chunks [k0..k0+7] and [k0+16..k0+23];
//    lanes 16-31 -> same rows, K chunks [k0+8..k0+15] and [k0+24..k0+31].
__device__ __forceinline__ v16h load_a(const _Float16* __restrict__ A, int lda,
                                       int m0, int k0, int lane) {
  int m  = m0 + (lane & 15);
  int kb = k0 + ((lane >> 4) << 3);
  const _Float16* p = A + (size_t)m * lda + kb;
  v16h a;
#pragma unroll
  for (int i = 0; i < 8; ++i) a[i] = p[i];
#pragma unroll
  for (int i = 0; i < 8; ++i) a[8 + i] = p[16 + i];
  return a;
}
__device__ __forceinline__ v16h load_a(const float* __restrict__ A, int lda,
                                       int m0, int k0, int lane) {
  int m  = m0 + (lane & 15);
  int kb = k0 + ((lane >> 4) << 3);
  const float* p = A + (size_t)m * lda + kb;
  v16h a;
#pragma unroll
  for (int i = 0; i < 8; ++i) a[i] = (_Float16)p[i];
#pragma unroll
  for (int i = 0; i < 8; ++i) a[8 + i] = (_Float16)p[16 + i];
  return a;
}

// B[k][n] = W[n][k] (computing X @ W^T, W row-major N x K, f16).
// Lane n holds col n; lanes 0-15: K chunks [k0..k0+7],[k0+16..k0+23]; lanes 16-31: [k0+8..k0+15],[k0+24..k0+31].
__device__ __forceinline__ v16h load_b_f16(const _Float16* __restrict__ W, int ldw,
                                           int n0, int k0, int lane) {
  int n  = n0 + (lane & 15);
  int kb = k0 + ((lane >> 4) << 3);
  const _Float16* p = W + (size_t)n * ldw + kb;
  v16h b;
#pragma unroll
  for (int i = 0; i < 8; ++i) b[i] = p[i];
#pragma unroll
  for (int i = 0; i < 8; ++i) b[8 + i] = p[16 + i];
  return b;
}

__device__ __forceinline__ void store_tile(float* __restrict__ Out, long long ldo,
                                           int mb, int n, float v) {
  Out[(size_t)mb * ldo + n] = v;
}
__device__ __forceinline__ void store_tile(_Float16* __restrict__ Out, long long ldo,
                                           int mb, int n, float v) {
  Out[(size_t)mb * ldo + n] = (_Float16)v;
}

// ---- wave-per-2-tiles GEMM: Out[M x N] = A[M x K] @ W[N x K]^T(f16) (+bias)(+tanh) ----
// One A fragment feeds two v_wmma per K step.  N must be a multiple of 32.
template <typename TA, typename TO>
__global__ __launch_bounds__(128)
void gemm16(const TA* __restrict__ A, int lda,
            const _Float16* __restrict__ W, int ldw,
            const float* __restrict__ bias,
            TO* __restrict__ Out, long long ldo,
            int N, int K, int act) {
  int lane = threadIdx.x;
  int tw   = blockIdx.x * blockDim.y + threadIdx.y;
  int n0   = tw * 32;
  if (n0 >= N) return;                 // wave-uniform exit
  int m0 = blockIdx.y * 16;

  v8f acc0 = {0.f, 0.f, 0.f, 0.f, 0.f, 0.f, 0.f, 0.f};
  v8f acc1 = {0.f, 0.f, 0.f, 0.f, 0.f, 0.f, 0.f, 0.f};
  const _Float16* pfb = W + (size_t)(n0 + (lane & 15)) * ldw;
  for (int k0 = 0; k0 < K; k0 += 32) {
    v16h a  = load_a(A, lda, m0, k0, lane);
    v16h b0 = load_b_f16(W, ldw, n0,      k0, lane);
    v16h b1 = load_b_f16(W, ldw, n0 + 16, k0, lane);
    acc0 = __builtin_amdgcn_wmma_f32_16x16x32_f16(false, a, false, b0,
                                                  (short)0, acc0, false, false);
    acc1 = __builtin_amdgcn_wmma_f32_16x16x32_f16(false, a, false, b1,
                                                  (short)0, acc1, false, false);
    if (k0 + 128 < K) __builtin_prefetch(pfb + k0 + 128, 0, 0);  // global_prefetch_b8
  }
  int na = n0 + (lane & 15), nb = na + 16;
  int mb = m0 + ((lane >> 4) << 3);
  float bva = bias ? bias[na] : 0.f;
  float bvb = bias ? bias[nb] : 0.f;
#pragma unroll
  for (int r = 0; r < 8; ++r) {
    float v0 = acc0[r] + bva;
    float v1 = acc1[r] + bvb;
    if (act == 1) { v0 = tanhf(v0); v1 = tanhf(v1); }
    store_tile(Out, ldo, mb + r, na, v0);
    store_tile(Out, ldo, mb + r, nb, v1);
  }
}

// ---- fused LSTM cell: 4 waves compute i,f,g,o gate tiles (WMMA), pointwise in LDS ----
// All matrix operands f16; cell state kept fp32 (+f16 shadow for downstream GEMMs).
__global__ __launch_bounds__(128)
void lstm_cell_wmma(const _Float16* __restrict__ X16, int ldx, int Ed,
                    const _Float16* __restrict__ Hin16, int ldhi,
                    const float* __restrict__ Cin, int ldci,
                    const _Float16* __restrict__ Wih,   // (4H x Ed)
                    const _Float16* __restrict__ Whh,   // (4H x Hd)
                    const float* __restrict__ bih, const float* __restrict__ bhh,
                    int Hd,
                    _Float16* __restrict__ Hout16, int ldho,
                    float* __restrict__ Cout, _Float16* __restrict__ Cout16, int ldco,
                    float* __restrict__ Hcopy, long long ldhc) {
  __shared__ float g[4][16][16];
  int lane = threadIdx.x;
  int gate = threadIdx.y;           // 0..3
  int n0   = blockIdx.x * 16;       // hidden column block

  const _Float16* Wg = Wih + (size_t)(gate * Hd) * Ed;
  const _Float16* Ug = Whh + (size_t)(gate * Hd) * Hd;

  v8f acc = {0.f, 0.f, 0.f, 0.f, 0.f, 0.f, 0.f, 0.f};
  for (int k0 = 0; k0 < Ed; k0 += 32) {
    v16h a = load_a(X16, ldx, 0, k0, lane);
    v16h b = load_b_f16(Wg, Ed, n0, k0, lane);
    acc = __builtin_amdgcn_wmma_f32_16x16x32_f16(false, a, false, b,
                                                 (short)0, acc, false, false);
  }
  for (int k0 = 0; k0 < Hd; k0 += 32) {
    v16h a = load_a(Hin16, ldhi, 0, k0, lane);
    v16h b = load_b_f16(Ug, Hd, n0, k0, lane);
    acc = __builtin_amdgcn_wmma_f32_16x16x32_f16(false, a, false, b,
                                                 (short)0, acc, false, false);
  }
  int n  = lane & 15;
  int mb = (lane >> 4) << 3;
  float bv = bih[gate * Hd + n0 + n] + bhh[gate * Hd + n0 + n];
#pragma unroll
  for (int r = 0; r < 8; ++r) g[gate][mb + r][n] = acc[r] + bv;
  __syncthreads();

  int tid = threadIdx.y * 32 + lane;   // 0..127
  for (int e = tid; e < 256; e += 128) {
    int m = e >> 4, nn = e & 15;
    float iv = sigm(g[0][m][nn]);
    float fv = sigm(g[1][m][nn]);
    float gv = tanhf(g[2][m][nn]);
    float ov = sigm(g[3][m][nn]);
    float c  = fv * Cin[(size_t)m * ldci + n0 + nn] + iv * gv;
    float h  = ov * tanhf(c);
    Cout[(size_t)m * ldco + n0 + nn]   = c;
    Cout16[(size_t)m * ldco + n0 + nn] = (_Float16)c;
    Hout16[(size_t)m * ldho + n0 + nn] = (_Float16)h;
    if (Hcopy) Hcopy[(size_t)m * ldhc + n0 + nn] = h;
  }
}

// ---- small utility kernels ----
__global__ void f32_to_f16_k(const float* __restrict__ s, _Float16* __restrict__ d, long long n) {
  long long i = (long long)blockIdx.x * blockDim.x + threadIdx.x;
  long long stride = (long long)gridDim.x * blockDim.x;
  for (; i < n; i += stride) d[i] = (_Float16)s[i];
}
__global__ void zero_f32_k(float* p, int n) {
  int i = blockIdx.x * blockDim.x + threadIdx.x;
  if (i < n) p[i] = 0.f;
}
__global__ void zero_f16_k(_Float16* p, int n) {
  int i = blockIdx.x * blockDim.x + threadIdx.x;
  if (i < n) p[i] = (_Float16)0.f;
}
__global__ void gather_rows_k(const float* __restrict__ emb, const int* __restrict__ idx,
                              float* __restrict__ out, _Float16* __restrict__ out16, int D) {
  int r = blockIdx.x;
  const float* src = emb + (size_t)idx[r] * D;
  for (int j = threadIdx.x; j < D; j += blockDim.x) {
    float v = src[j];
    out[(size_t)r * D + j] = v;
    out16[(size_t)r * D + j] = (_Float16)v;
  }
}
__global__ void gather_topic_k(const float* __restrict__ emb, const int* __restrict__ topic,
                               _Float16* __restrict__ xtb16) {
  int t = blockIdx.x, b = blockIdx.y;
  int tok = topic[b * TTs + t];
  for (int j = threadIdx.x; j < Es; j += blockDim.x)
    xtb16[((size_t)t * BDIM + b) * Es + j] = (_Float16)emb[(size_t)tok * Es + j];
}
__global__ void gather_essay_k(const float* __restrict__ emb, const int* __restrict__ essay,
                               _Float16* __restrict__ embSeq16) {
  int s = blockIdx.x;   // 0..64 ; [0]=emb(tok0), [s+1]=emb(tok for step s)
  int b = blockIdx.y;
  int col = (s == 0) ? 0 : (s < Ts ? s : Ts - 1);
  int tok = essay[b * Ts + col];
  for (int j = threadIdx.x; j < Es; j += blockDim.x)
    embSeq16[((size_t)s * BDIM + b) * Es + j] = (_Float16)emb[(size_t)tok * Es + j];
}
__global__ void copy_emb_k(const _Float16* __restrict__ emb16, _Float16* __restrict__ dec_in16) {
  int b = blockIdx.x;
  for (int j = threadIdx.x; j < Es; j += blockDim.x)
    dec_in16[(size_t)b * INs + j] = emb16[(size_t)b * Es + j];
}

// memory-network pointwise: M = gate*tanh(MU1+eV1) + (1-gate)*M ; sanitize non-finite
__global__ void mem_update_k(float* __restrict__ M, _Float16* __restrict__ M16,
                             const float* __restrict__ MU1, const float* __restrict__ MU2,
                             const float* __restrict__ eV1, const float* __restrict__ eV2) {
  int i = blockIdx.x * blockDim.x + threadIdx.x;
  if (i >= BDIM * Ls * Es) return;
  int j = i & (Es - 1);
  int b = i / (Ls * Es);
  float md = M[i];
  float mt = tanhf(MU1[i] + eV1[b * Es + j]);
  float gt = sigm(MU2[i] + eV2[b * Es + j]);
  float r  = gt * mt + (1.f - gt) * md;
  if (!(fabsf(r) <= 3.4e38f)) r = 1e-31f;   // catches NaN/Inf
  M[i] = r;
  M16[i] = (_Float16)r;
}

// topic attention: alpha = softmax_t(pre_t . q); c_t = sum_t alpha*topics -> dec_in[512:1536]
__global__ void attn_topic_k(const float* __restrict__ pre_t, const float* __restrict__ q,
                             const float* __restrict__ topics, _Float16* __restrict__ dec_in16) {
  int b = blockIdx.x, tid = threadIdx.x;
  __shared__ float red[256];
  __shared__ float sc[TTs];
  for (int t = 0; t < TTs; ++t) {
    float p = 0.f;
    for (int a = tid; a < As; a += 256)
      p += pre_t[((size_t)b * TTs + t) * As + a] * q[(size_t)b * As + a];
    red[tid] = p; __syncthreads();
    for (int o = 128; o > 0; o >>= 1) { if (tid < o) red[tid] += red[tid + o]; __syncthreads(); }
    if (tid == 0) sc[t] = red[0];
    __syncthreads();
  }
  float mx = sc[0];
  for (int t = 1; t < TTs; ++t) mx = fmaxf(mx, sc[t]);
  float al[TTs]; float sum = 0.f;
  for (int t = 0; t < TTs; ++t) { al[t] = __expf(sc[t] - mx); sum += al[t]; }
  float inv = 1.f / sum;
  for (int j = tid; j < EOs; j += 256) {
    float acc = 0.f;
    for (int t = 0; t < TTs; ++t)
      acc += al[t] * inv * topics[((size_t)b * TTs + t) * EOs + j];
    dec_in16[(size_t)b * INs + Es + j] = (_Float16)acc;
  }
}

// memory attention: qm = softmax_l(M . v); m_t = sum_l qm*M -> dec_in[1536:2048]
__global__ void attn_mem_k(const float* __restrict__ M, const float* __restrict__ v,
                           _Float16* __restrict__ dec_in16) {
  int b = blockIdx.x, tid = threadIdx.x;
  int lane = tid & 31, wv = tid >> 5;   // 8 waves
  __shared__ float sc[Ls];
  for (int l = wv; l < Ls; l += 8) {
    const float* mr = M + ((size_t)b * Ls + l) * Es;
    const float* vr = v + (size_t)b * Es;
    float p = 0.f;
    for (int a = lane; a < Es; a += 32) p += mr[a] * vr[a];
    for (int o = 16; o > 0; o >>= 1) p += __shfl_down(p, o, 32);
    if (lane == 0) sc[l] = p;
  }
  __syncthreads();
  float mx = -1e30f;
  for (int l = 0; l < Ls; ++l) mx = fmaxf(mx, sc[l]);
  float sum = 0.f;
  for (int l = 0; l < Ls; ++l) sum += __expf(sc[l] - mx);
  float inv = 1.f / sum;
  __syncthreads();
  if (tid < Ls) sc[tid] = __expf(sc[tid] - mx) * inv;
  __syncthreads();
  for (int j = tid; j < Es; j += 256) {
    float acc = 0.f;
    for (int l = 0; l < Ls; ++l) acc += sc[l] * M[((size_t)b * Ls + l) * Es + j];
    dec_in16[(size_t)b * INs + EOs + Es + j] = (_Float16)acc;
  }
}

extern "C" void kernel_launch(void* const* d_in, const int* in_sizes, int n_in,
                              void* d_out, int out_size, void* d_ws, size_t ws_size,
                              hipStream_t stream) {
  const float* embedding = (const float*)d_in[0];
  const float* encWihF = (const float*)d_in[1];
  const float* encWhhF = (const float*)d_in[2];
  const float* encBihF = (const float*)d_in[3];
  const float* encBhhF = (const float*)d_in[4];
  const float* encWihB = (const float*)d_in[5];
  const float* encWhhB = (const float*)d_in[6];
  const float* encBihB = (const float*)d_in[7];
  const float* encBhhB = (const float*)d_in[8];
  const float* W1  = (const float*)d_in[9];
  const float* W2  = (const float*)d_in[10];
  const float* memW = (const float*)d_in[11];
  const float* memB = (const float*)d_in[12];
  const float* U1 = (const float*)d_in[13];
  const float* V1 = (const float*)d_in[14];
  const float* U2 = (const float*)d_in[15];
  const float* V2 = (const float*)d_in[16];
  const float* sharpW = (const float*)d_in[17];
  const float* sharpB = (const float*)d_in[18];
  const float* decWih = (const float*)d_in[19];
  const float* decWhh = (const float*)d_in[20];
  const float* decBih = (const float*)d_in[21];
  const float* decBhh = (const float*)d_in[22];
  const float* fcW = (const float*)d_in[23];
  const float* fcB = (const float*)d_in[24];
  const int* topic = (const int*)d_in[25];
  const int* essay = (const int*)d_in[27];
  const int* mems  = (const int*)d_in[29];
  float* out = (float*)d_out;

  // ---- workspace carve-up (256B aligned) ----
  char* wp = (char*)d_ws;
  auto alloc = [&](size_t bytes) -> char* {
    char* r = wp; wp += (bytes + 255) & ~(size_t)255; return r;
  };
  auto allocH = [&](size_t n) -> _Float16* { return (_Float16*)alloc(n * sizeof(_Float16)); };
  auto allocF = [&](size_t n) -> float*    { return (float*)alloc(n * sizeof(float)); };

  _Float16* fcWh     = allocH((size_t)Vs * EOs);
  _Float16* decWihH  = allocH((size_t)4 * EOs * TEMPs);
  _Float16* decWhhH  = allocH((size_t)4 * EOs * EOs);
  _Float16* sharpWh  = allocH((size_t)TEMPs * INs);
  _Float16* U1h = allocH((size_t)Es * Es);
  _Float16* V1h = allocH((size_t)Es * Es);
  _Float16* U2h = allocH((size_t)Es * Es);
  _Float16* V2h = allocH((size_t)Es * Es);
  _Float16* W1h  = allocH((size_t)As * EOs);
  _Float16* W2h  = allocH((size_t)As * EOs);
  _Float16* memWh = allocH((size_t)Es * EOs);
  _Float16* eWihFh = allocH((size_t)4 * Hs * Es);
  _Float16* eWhhFh = allocH((size_t)4 * Hs * Hs);
  _Float16* eWihBh = allocH((size_t)4 * Hs * Es);
  _Float16* eWhhBh = allocH((size_t)4 * Hs * Hs);

  _Float16* xtb16    = allocH((size_t)TTs * BDIM * Es);
  float*    topics   = allocF((size_t)BDIM * TTs * EOs);
  float*    pre_t    = allocF((size_t)BDIM * TTs * As);
  float*    Mbuf     = allocF((size_t)BDIM * Ls * Es);
  _Float16* M16      = allocH((size_t)BDIM * Ls * Es);
  float*    MU1      = allocF((size_t)BDIM * Ls * Es);
  float*    MU2      = allocF((size_t)BDIM * Ls * Es);
  float*    eV1      = allocF((size_t)BDIM * Es);
  float*    eV2      = allocF((size_t)BDIM * Es);
  _Float16* embSeq16 = allocH((size_t)(Ts + 1) * BDIM * Es);
  _Float16* h16A = allocH((size_t)BDIM * EOs);
  _Float16* h16B = allocH((size_t)BDIM * EOs);
  float*    cA   = allocF((size_t)BDIM * EOs);
  float*    cB   = allocF((size_t)BDIM * EOs);
  _Float16* c16  = allocH((size_t)BDIM * EOs);
  _Float16* decin16 = allocH((size_t)BDIM * INs);
  _Float16* xs16    = allocH((size_t)BDIM * TEMPs);
  float*    qbuf = allocF((size_t)BDIM * As);
  float*    vbuf = allocF((size_t)BDIM * Es);

  auto cvt = [&](const float* s, _Float16* d, long long n) {
    int grid = (int)((n + 1023) / 1024); if (grid > 4096) grid = 4096;
    f32_to_f16_k<<<grid, 256, 0, stream>>>(s, d, n);
  };
  auto gemmHF = [&](const _Float16* A, int lda, const _Float16* W, int ldw, const float* bias,
                    float* Out, long long ldo, int Mrows, int N, int K, int act) {
    dim3 blk(32, 4), grd((N / 32 + 3) / 4, Mrows / 16);
    gemm16<_Float16, float><<<grd, blk, 0, stream>>>(A, lda, W, ldw, bias, Out, ldo, N, K, act);
  };
  auto gemmHH = [&](const _Float16* A, int lda, const _Float16* W, int ldw, const float* bias,
                    _Float16* Out, long long ldo, int Mrows, int N, int K, int act) {
    dim3 blk(32, 4), grd((N / 32 + 3) / 4, Mrows / 16);
    gemm16<_Float16, _Float16><<<grd, blk, 0, stream>>>(A, lda, W, ldw, bias, Out, ldo, N, K, act);
  };
  auto gemmFF = [&](const float* A, int lda, const _Float16* W, int ldw, const float* bias,
                    float* Out, long long ldo, int Mrows, int N, int K, int act) {
    dim3 blk(32, 4), grd((N / 32 + 3) / 4, Mrows / 16);
    gemm16<float, float><<<grd, blk, 0, stream>>>(A, lda, W, ldw, bias, Out, ldo, N, K, act);
  };

  // ---- phase 0: f16 weight shadows (once per launch; deterministic) ----
  cvt(fcW, fcWh, (long long)Vs * EOs);
  cvt(decWih, decWihH, (long long)4 * EOs * TEMPs);
  cvt(decWhh, decWhhH, (long long)4 * EOs * EOs);
  cvt(sharpW, sharpWh, (long long)TEMPs * INs);
  cvt(U1, U1h, (long long)Es * Es);  cvt(V1, V1h, (long long)Es * Es);
  cvt(U2, U2h, (long long)Es * Es);  cvt(V2, V2h, (long long)Es * Es);
  cvt(W1, W1h, (long long)As * EOs); cvt(W2, W2h, (long long)As * EOs);
  cvt(memW, memWh, (long long)Es * EOs);
  cvt(encWihF, eWihFh, (long long)4 * Hs * Es); cvt(encWhhF, eWhhFh, (long long)4 * Hs * Hs);
  cvt(encWihB, eWihBh, (long long)4 * Hs * Es); cvt(encWhhB, eWhhBh, (long long)4 * Hs * Hs);

  // ---- phase 1: gathers ----
  gather_topic_k<<<dim3(TTs, BDIM), 256, 0, stream>>>(embedding, topic, xtb16);
  gather_rows_k<<<BDIM * Ls, 256, 0, stream>>>(embedding, mems, Mbuf, M16, Es);
  gather_essay_k<<<dim3(Ts + 1, BDIM), 256, 0, stream>>>(embedding, essay, embSeq16);

  // ---- phase 2: bidirectional encoder (ping-pong h16/c; fwd cols 0..511, bwd 512..1023) ----
  zero_f16_k<<<(BDIM * EOs + 255) / 256, 256, 0, stream>>>(h16A, BDIM * EOs);
  zero_f32_k<<<(BDIM * EOs + 255) / 256, 256, 0, stream>>>(cA, BDIM * EOs);
  _Float16 *hp16 = h16A, *hn16 = h16B;
  float *cp = cA, *cn = cB;
  for (int t = 0; t < TTs; ++t) {
    // forward
    lstm_cell_wmma<<<dim3(Hs / 16), dim3(32, 4), 0, stream>>>(
        xtb16 + (size_t)t * BDIM * Es, Es, Es, hp16, EOs, cp, EOs,
        eWihFh, eWhhFh, encBihF, encBhhF, Hs,
        hn16, EOs, cn, c16, EOs, topics + (size_t)t * EOs, (long long)TTs * EOs);
    // backward (consumes reversed input, writes reversed slot, high half)
    int tb = TTs - 1 - t;
    lstm_cell_wmma<<<dim3(Hs / 16), dim3(32, 4), 0, stream>>>(
        xtb16 + (size_t)tb * BDIM * Es, Es, Es, hp16 + Hs, EOs, cp + Hs, EOs,
        eWihBh, eWhhBh, encBihB, encBhhB, Hs,
        hn16 + Hs, EOs, cn + Hs, c16 + Hs, EOs,
        topics + (size_t)tb * EOs + Hs, (long long)TTs * EOs);
    _Float16* th = hp16; hp16 = hn16; hn16 = th;
    float* tc = cp; cp = cn; cn = tc;
  }
  // final encoder state: hp16 (f16), cp (f32); c16 holds final c in f16

  // pre_t = topics @ W1^T  (128 x 1024 -> 128 x 512), one-shot fp32-A path
  gemmFF(topics, EOs, W1h, EOs, nullptr, pre_t, As, BDIM * TTs, As, EOs, 0);

  // ---- phase 3: memory update + before_feed helpers ----
  auto mem_step = [&](const _Float16* embS16) {
    gemmHF(embS16, Es, V1h, Es, nullptr, eV1, Es, BDIM, Es, Es, 0);
    gemmHF(embS16, Es, V2h, Es, nullptr, eV2, Es, BDIM, Es, Es, 0);
    gemmHF(M16, Es, U1h, Es, nullptr, MU1, Es, BDIM * Ls, Es, Es, 0);
    gemmHF(M16, Es, U2h, Es, nullptr, MU2, Es, BDIM * Ls, Es, Es, 0);
    mem_update_k<<<(BDIM * Ls * Es + 255) / 256, 256, 0, stream>>>(Mbuf, M16, MU1, MU2, eV1, eV2);
  };
  auto before_feed = [&](const _Float16* embS16, const _Float16* h16cur, const _Float16* c16cur) {
    gemmHF(c16cur, EOs, W2h, EOs, nullptr, qbuf, As, BDIM, As, EOs, 0);
    attn_topic_k<<<BDIM, 256, 0, stream>>>(pre_t, qbuf, topics, decin16);
    gemmHF(h16cur, EOs, memWh, EOs, memB, vbuf, Es, BDIM, Es, EOs, 1 /*tanh*/);
    attn_mem_k<<<BDIM, 256, 0, stream>>>(Mbuf, vbuf, decin16);
    copy_emb_k<<<BDIM, 256, 0, stream>>>(embS16, decin16);
  };

  mem_step(embSeq16);                  // emb0
  before_feed(embSeq16, hp16, c16);    // dec_in for step 0

  // ---- phase 4: decoder, 64 sequential steps ----
  for (int s = 0; s < Ts; ++s) {
    // xs = dec_in @ sharp_W^T + sharp_b  (f16 out, feeds LSTM WMMA directly)
    gemmHH(decin16, INs, sharpWh, INs, sharpB, xs16, TEMPs, BDIM, TEMPs, INs, 0);
    // LSTM cell (EO=1024): read hp16/cp, write hn16/cn (+c16 shadow)
    lstm_cell_wmma<<<dim3(EOs / 16), dim3(32, 4), 0, stream>>>(
        xs16, TEMPs, TEMPs, hp16, EOs, cp, EOs,
        decWihH, decWhhH, decBih, decBhh, EOs,
        hn16, EOs, cn, c16, EOs, nullptr, 0);
    // logits[:, s, :] = h @ fc_W^T + fc_b   (16 x 32000, K=1024; fcWh L2-resident)
    gemmHF(hn16, EOs, fcWh, EOs, fcB, out + (size_t)s * Vs, (long long)Ts * Vs,
           BDIM, Vs, EOs, 0);
    // memory update + next dec_in with this step's token embedding
    const _Float16* embS16 = embSeq16 + (size_t)(s + 1) * BDIM * Es;
    mem_step(embS16);
    before_feed(embS16, hn16, c16);
    _Float16* th = hp16; hp16 = hn16; hn16 = th;
    float* tc = cp; cp = cn; cn = tc;
  }
  (void)in_sizes; (void)n_in; (void)out_size; (void)ws_size;
}
